// FrequencyAdaptiveSpatialMix_64639257805135
// MI455X (gfx1250) — matrix operand order repaired and checked by reading, not compiled, not run on hardware
//
#include <hip/hip_runtime.h>
#include <hip/hip_bf16.h>

// ---------------------------------------------------------------------------
// FrequencyAdaptiveSpatialMix for MI455X (gfx1250), wave32.
// GEMMs: bf16 WMMA (v_wmma_f32_16x16x32_bf16), TDM (tensor_load_to_lds) staging.
// B=8, H=W=64, T=4096, C=768.
// ---------------------------------------------------------------------------

typedef __attribute__((ext_vector_type(16))) __bf16 v16bf;
typedef __attribute__((ext_vector_type(8)))  float  v8f;
typedef __attribute__((ext_vector_type(4)))  unsigned int u32x4;
typedef __attribute__((ext_vector_type(8)))  int    i32x8;
typedef __attribute__((ext_vector_type(4)))  int    i32x4;
typedef unsigned short ushort_t;

#define BDIM   8
#define HDIM   64
#define WDIM   64
#define TDIM   4096
#define CDIM   768
#define MDIM   (BDIM * TDIM)   // 32768
#define NCH    32              // wkv chunks
#define CLEN   (TDIM / NCH)    // 128 steps per chunk
#define NEGINF (-1e38f)

#if __has_builtin(__builtin_amdgcn_tensor_load_to_lds) && \
    __has_builtin(__builtin_amdgcn_s_wait_tensorcnt)
#define USE_TDM 1
#else
#define USE_TDM 0
#endif

__device__ __forceinline__ ushort_t f2bf(float f) {
    union { float f; unsigned int u; } c; c.f = f;
    unsigned int u = c.u;
    u += 0x7FFFu + ((u >> 16) & 1u);   // round to nearest even
    return (ushort_t)(u >> 16);
}

// ---------------------------------------------------------------------------
// 1) gctx = mean_T(x)   (atomic partials over 64 T-segments)
// ---------------------------------------------------------------------------
__global__ void zero_f32_kernel(float* p, int n) {
    int i = blockIdx.x * blockDim.x + threadIdx.x;
    if (i < n) p[i] = 0.0f;
}

__global__ __launch_bounds__(256) void gctx_kernel(const float* __restrict__ x,
                                                   float* __restrict__ gctx) {
    int b = blockIdx.x, seg = blockIdx.y, tid = threadIdx.x;
    float a0 = 0.f, a1 = 0.f, a2 = 0.f;
    for (int i = 0; i < TDIM / 64; ++i) {
        const float* row = x + ((size_t)b * TDIM + seg * (TDIM / 64) + i) * CDIM;
        a0 += row[tid]; a1 += row[tid + 256]; a2 += row[tid + 512];
    }
    const float s = 1.0f / (float)TDIM;
    atomicAdd(&gctx[b * CDIM + tid],       a0 * s);
    atomicAdd(&gctx[b * CDIM + tid + 256], a1 * s);
    atomicAdd(&gctx[b * CDIM + tid + 512], a2 * s);
}

// ---------------------------------------------------------------------------
// 2) h = tanh(gctx@Wd1+bd1); mod = sigmoid(h@Wd2+bd2); w = decay + mean_B(mod)
// ---------------------------------------------------------------------------
__global__ __launch_bounds__(256) void decay_kernel(
    const float* __restrict__ gctx, const float* __restrict__ Wd1,
    const float* __restrict__ bd1,  const float* __restrict__ Wd2,
    const float* __restrict__ bd2,  const float* __restrict__ spatial_decay,
    float* __restrict__ wvec) {
    __shared__ float sg[BDIM * CDIM];
    __shared__ float sh[BDIM * (CDIM / 4)];
    int tid = threadIdx.x;
    for (int i = tid; i < BDIM * CDIM; i += 256) sg[i] = gctx[i];
    __syncthreads();
    for (int o = tid; o < BDIM * (CDIM / 4); o += 256) {
        int b = o / (CDIM / 4), j = o % (CDIM / 4);
        float s = bd1[j];
        for (int c = 0; c < CDIM; ++c) s += sg[b * CDIM + c] * Wd1[c * (CDIM / 4) + j];
        sh[o] = tanhf(s);
    }
    __syncthreads();
    for (int c = tid; c < CDIM; c += 256) {
        float acc = 0.f;
        for (int b = 0; b < BDIM; ++b) {
            float s = bd2[c];
            for (int j = 0; j < CDIM / 4; ++j) s += sh[b * (CDIM / 4) + j] * Wd2[j * CDIM + c];
            acc += 1.0f / (1.0f + __expf(-s));
        }
        wvec[c] = spatial_decay[c] + acc * (1.0f / (float)BDIM);
    }
}

// ---------------------------------------------------------------------------
// 3) Weight convert f32 -> bf16, transposed to [n][k] (k contiguous).
// ---------------------------------------------------------------------------
__global__ void wconv_kernel(const float* __restrict__ Wk, const float* __restrict__ Wv,
                             const float* __restrict__ Wr, const float* __restrict__ Wo,
                             ushort_t* __restrict__ WT) {
    int idx = blockIdx.x * blockDim.x + threadIdx.x;
    const int per = CDIM * CDIM;
    if (idx >= 4 * per) return;
    int mat = idx / per, rem = idx % per;
    int k = rem / CDIM, n = rem % CDIM;
    const float* W = (mat == 0) ? Wk : (mat == 1) ? Wv : (mat == 2) ? Wr : Wo;
    WT[(size_t)mat * per + (size_t)n * CDIM + k] = f2bf(W[rem]);
}

// ---------------------------------------------------------------------------
// 4) q_shift + token mix -> Xk/Xv/Xr in bf16
// ---------------------------------------------------------------------------
__global__ __launch_bounds__(256) void qshift_mix_kernel(
    const float* __restrict__ x,
    const float* __restrict__ mix_k, const float* __restrict__ mix_v,
    const float* __restrict__ mix_r,
    ushort_t* __restrict__ Xk, ushort_t* __restrict__ Xv, ushort_t* __restrict__ Xr) {
    int idx = blockIdx.x * blockDim.x + threadIdx.x;
    if (idx >= MDIM * CDIM) return;
    int c  = idx % CDIM;
    int bt = idx / CDIM;
    int t  = bt % TDIM;
    int h  = t / WDIM, wcol = t % WDIM;
    const int g = CDIM / 4;
    float xx = 0.0f;
    if (c < g)            { if (wcol > 0)        xx = x[idx - CDIM]; }        // (h, w-1)
    else if (c < 2 * g)   { if (wcol < WDIM - 1) xx = x[idx + CDIM]; }        // (h, w+1)
    else if (c < 3 * g)   { if (h > 0)           xx = x[idx - WDIM * CDIM]; } // (h-1, w)
    else                  { if (h < HDIM - 1)    xx = x[idx + WDIM * CDIM]; } // (h+1, w)
    float xs = x[idx];
    float mk = mix_k[c], mv = mix_v[c], mr = mix_r[c];
    Xk[idx] = f2bf(xs * mk + xx * (1.0f - mk));
    Xv[idx] = f2bf(xs * mv + xx * (1.0f - mv));
    Xr[idx] = f2bf(xs * mr + xx * (1.0f - mr));
}

// ---------------------------------------------------------------------------
// 5) WMMA bf16 GEMM with TDM staging.
//    128x128 block tile, 8 waves (2x4), wave = 64x32 = 4x2 wmma 16x16 tiles.
//    KC=32 per step, double-buffered LDS; row stride padded to 80B by the
//    TDM pad feature (16B pad every 16 DWORDs) -> conflict-free ds_load_b128.
// ---------------------------------------------------------------------------
#define GBM 128
#define GBN 128
#define GKC 32
#define LDT 40   // padded LDS row stride in ushorts (80B)

union FragAB { uint4 q[2]; v16bf v; };

#if USE_TDM
// One TENSOR_LOAD_TO_LDS moves a 128x32 bf16 tile (row stride = Kdim elems)
// into LDS with a 16B pad after each 64B row -> 80B LDS row stride.
__device__ __forceinline__ void tdm_load_tile(unsigned int lds_byte_addr,
                                              const ushort_t* gsrc, int kdim) {
    unsigned long long ga = (unsigned long long)gsrc;
    u32x4 g0;
    g0[0] = 1u;                                                  // count=1, no gather
    g0[1] = lds_byte_addr;                                       // D#.lds_addr
    g0[2] = (unsigned int)ga;                                    // global_addr[31:0]
    g0[3] = ((unsigned int)(ga >> 32) & 0x01FFFFFFu) | (2u << 30); // addr[56:32] | type=2
    i32x8 g1;
    g1[0] = (1 << 16)            // data_size = 2 bytes (bf16)
          | (1 << 20)            // pad_enable
          | (3 << 22)            // pad_interval: 16 DWORDs (= one 64B row)
          | (3 << 25);           // pad_amount:   4 DWORDs (16B)
    g1[1] = (kdim & 0xFFFF) << 16;          // tensor_dim0[15:0] (=768)
    g1[2] = (int)0x80000000;                // tensor_dim0 hi=0 | tensor_dim1 lo (32768)
    g1[3] = (GKC << 16);                    // tensor_dim1 hi=0 | tile_dim0=32
    g1[4] = GBM;                            // tile_dim1=128 rows | tile_dim2=0
    g1[5] = kdim;                           // tensor_dim0_stride lo (=768)
    g1[6] = 0;                              // stride0 hi | stride1 lo
    g1[7] = 0;                              // stride1 hi
    i32x4 gz4 = {0, 0, 0, 0};               // groups 2/3: 2-D tensor
    i32x8 gz8 = {0, 0, 0, 0, 0, 0, 0, 0};   // extra group (6-arg toolchain form)
    __builtin_amdgcn_tensor_load_to_lds(g0, g1, gz4, gz4, gz8, 0);
}
#endif

__global__ __launch_bounds__(256) void gemm_bf16_kernel(
    const ushort_t* __restrict__ Xall, const ushort_t* __restrict__ WTall,
    float* __restrict__ Oall, int Mdim, int Kdim, int Ndim, int sig_z) {
    const int bm = blockIdx.x * GBM;
    const int bn = blockIdx.y * GBN;
    const int z  = blockIdx.z;
    const ushort_t* __restrict__ X  = Xall  + (size_t)z * Mdim * Kdim;
    const ushort_t* __restrict__ WT = WTall + (size_t)z * Ndim * Kdim;
    float* __restrict__ O           = Oall  + (size_t)z * Mdim * Ndim;

    __shared__ ushort_t lA[2][GBM * LDT];
    __shared__ ushort_t lB[2][GBN * LDT];

    const int tid  = threadIdx.x;
    const int lane = tid & 31;
    const int wid  = tid >> 5;
    const int wm   = (wid >> 2) * 64;   // 0 / 64
    const int wn   = (wid & 3) * 32;    // 0,32,64,96

    v8f acc[4][2];
#pragma unroll
    for (int i = 0; i < 4; ++i)
#pragma unroll
        for (int j = 0; j < 2; ++j) {
            v8f zz = {0.f, 0.f, 0.f, 0.f, 0.f, 0.f, 0.f, 0.f};
            acc[i][j] = zz;
        }

    const int NK = Kdim / GKC;
    const ushort_t* gA0 = &X [(size_t)bm * Kdim];
    const ushort_t* gB0 = &WT[(size_t)bn * Kdim];

    // fragment addressing (16-bit WMMA layouts):
    //   A: lane m = lane&15; k-halves {0..7,16..23} (lanes 0-15) / {8..15,24..31}
    //   B: lane n = lane&15; k-halves {0..7,16..23} (lanes 0-15) / {8..15,24..31}
    const int khA = (lane >> 4) << 3;    // 0 or 8
    const int khB = (lane >> 4) << 4;    // 0 or 16

    auto compute = [&](int buf) {
        FragAB fa[4], fb[2];
#pragma unroll
        for (int j = 0; j < 2; ++j) {
            const ushort_t* pb = &lB[buf][(wn + j * 16 + (lane & 15)) * LDT + khA];
            fb[j].q[0] = *(const uint4*)pb;
            fb[j].q[1] = *(const uint4*)(pb + 16);
        }
#pragma unroll
        for (int i = 0; i < 4; ++i) {
            const ushort_t* pa = &lA[buf][(wm + i * 16 + (lane & 15)) * LDT + khA];
            fa[i].q[0] = *(const uint4*)pa;
            fa[i].q[1] = *(const uint4*)(pa + 16);
        }
#pragma unroll
        for (int i = 0; i < 4; ++i)
#pragma unroll
            for (int j = 0; j < 2; ++j)
                acc[i][j] = __builtin_amdgcn_wmma_f32_16x16x32_bf16(
                    false, fa[i].v, false, fb[j].v, (short)0, acc[i][j], false, false);
    };

#if USE_TDM
    if (wid == 0) {
        tdm_load_tile((unsigned int)(unsigned long long)&lA[0][0], gA0, Kdim);
        tdm_load_tile((unsigned int)(unsigned long long)&lB[0][0], gB0, Kdim);
        __builtin_amdgcn_s_wait_tensorcnt((unsigned short)0);
    }
    __syncthreads();
    for (int kc = 0; kc < NK; ++kc) {
        const int buf = kc & 1;
        if (kc + 1 < NK && wid == 0) {
            tdm_load_tile((unsigned int)(unsigned long long)&lA[buf ^ 1][0],
                          gA0 + (kc + 1) * GKC, Kdim);
            tdm_load_tile((unsigned int)(unsigned long long)&lB[buf ^ 1][0],
                          gB0 + (kc + 1) * GKC, Kdim);
        }
        compute(buf);
        if (kc + 1 < NK) {
            if (wid == 0) __builtin_amdgcn_s_wait_tensorcnt((unsigned short)0);
            __syncthreads();
        }
    }
#else
    // Fallback: synchronous staging, single buffer.
    for (int kc = 0; kc < NK; ++kc) {
        __syncthreads();
#pragma unroll
        for (int p = 0; p < 2; ++p) {
            int idx = tid + p * 256;
            int row = idx >> 2;
            int cc  = (idx & 3) << 3;
            uint4 va = *(const uint4*)&gA0[(size_t)row * Kdim + kc * GKC + cc];
            uint4 vb = *(const uint4*)&gB0[(size_t)row * Kdim + kc * GKC + cc];
            *(uint4*)&lA[0][row * LDT + cc] = va;
            *(uint4*)&lB[0][row * LDT + cc] = vb;
        }
        __syncthreads();
        compute(0);
    }
#endif

    // Epilogue: C/D layout — VGPR r: lanes 0-15 -> (M=r, N=lane); 16-31 -> (M=8+r, N=lane-16)
    const int do_sig = (z == sig_z);
    const int mb = bm + wm + ((lane >> 4) << 3);
    const int nb = bn + wn + (lane & 15);
#pragma unroll
    for (int i = 0; i < 4; ++i)
#pragma unroll
        for (int j = 0; j < 2; ++j)
#pragma unroll
            for (int r = 0; r < 8; ++r) {
                float val = acc[i][j][r];
                if (do_sig) val = 1.0f / (1.0f + __expf(-val));
                O[(size_t)(mb + i * 16 + r) * Ndim + nb + j * 16] = val;
            }
}

// ---------------------------------------------------------------------------
// 6) WKV chunked scan (uniform per-step decay => chunk transfer is affine
//    with log-shift w*CLEN).  id layout: ((chunk*B + b)*C + c), c fastest.
// ---------------------------------------------------------------------------
__global__ __launch_bounds__(256) void wkv_passA(
    const float* __restrict__ k, const float* __restrict__ v,
    const float* __restrict__ wvec,
    float* __restrict__ la, float* __restrict__ lb, float* __restrict__ lp) {
    int id = blockIdx.x * blockDim.x + threadIdx.x;
    if (id >= NCH * BDIM * CDIM) return;
    int c  = id % CDIM;
    int b  = (id / CDIM) % BDIM;
    int ch = id / (CDIM * BDIM);
    const float w = wvec[c];
    float a = 0.f, bs = 0.f, p = NEGINF;
    size_t base = ((size_t)b * TDIM + ch * CLEN) * CDIM + c;
    for (int i = 0; i < CLEN; ++i) {
        float kt = k[base + (size_t)i * CDIM];
        float vt = v[base + (size_t)i * CDIM];
        float ww2 = p + w;
        float p3  = fmaxf(ww2, kt);
        float f1  = __expf(ww2 - p3);
        float f2  = __expf(kt - p3);
        a  = f1 * a + f2 * vt;
        bs = f1 * bs + f2;
        p  = p3;
    }
    la[id] = a; lb[id] = bs; lp[id] = p;
}

__global__ __launch_bounds__(256) void wkv_passB(
    const float* __restrict__ la, const float* __restrict__ lb,
    const float* __restrict__ lp, const float* __restrict__ wvec,
    float* __restrict__ sa, float* __restrict__ sb, float* __restrict__ sp) {
    int id = blockIdx.x * blockDim.x + threadIdx.x;   // (b,c)
    if (id >= BDIM * CDIM) return;
    int c = id % CDIM;
    const float wL = wvec[c] * (float)CLEN;
    float a = 0.f, b = 0.f, p = NEGINF;
    for (int ch = 0; ch < NCH; ++ch) {
        int o = ch * BDIM * CDIM + id;
        sa[o] = a; sb[o] = b; sp[o] = p;
        float p1 = p + wL;
        float a2 = la[o], b2 = lb[o], p2 = lp[o];
        float pm = fmaxf(p1, p2);
        float e1 = __expf(p1 - pm), e2 = __expf(p2 - pm);
        a = e1 * a + e2 * a2;
        b = e1 * b + e2 * b2;
        p = pm;
    }
}

__global__ __launch_bounds__(256) void wkv_passC(
    const float* __restrict__ k, const float* __restrict__ v,
    const float* __restrict__ wvec, const float* __restrict__ u,
    const float* __restrict__ sa, const float* __restrict__ sb,
    const float* __restrict__ sp, float* __restrict__ y) {
    int id = blockIdx.x * blockDim.x + threadIdx.x;
    if (id >= NCH * BDIM * CDIM) return;
    int c  = id % CDIM;
    int b  = (id / CDIM) % BDIM;
    int ch = id / (CDIM * BDIM);
    const float w = wvec[c], uu = u[c];
    float a = sa[id], bs = sb[id], p = sp[id];
    size_t base = ((size_t)b * TDIM + ch * CLEN) * CDIM + c;
    for (int i = 0; i < CLEN; ++i) {
        float kt = k[base + (size_t)i * CDIM];
        float vt = v[base + (size_t)i * CDIM];
        float ww = uu + kt;
        float p2 = fmaxf(p, ww);
        float e1 = __expf(p - p2);
        float e2 = __expf(ww - p2);
        y[base + (size_t)i * CDIM] = (e1 * a + e2 * vt) / (e1 * bs + e2);
        float ww2 = p + w;
        float p3  = fmaxf(ww2, kt);
        float f1  = __expf(ww2 - p3);
        float f2  = __expf(kt - p3);
        a  = f1 * a + f2 * vt;
        bs = f1 * bs + f2;
        p  = p3;
    }
}

// ---------------------------------------------------------------------------
// 7) z = bf16( sr * LayerNorm(y) )   — one block (256 thr) per row of 768
// ---------------------------------------------------------------------------
__global__ __launch_bounds__(256) void ln_kernel(
    const float* __restrict__ y, const float* __restrict__ sr,
    const float* __restrict__ g, const float* __restrict__ bln,
    ushort_t* __restrict__ zout) {
    __shared__ float red[256];
    const int row = blockIdx.x, tid = threadIdx.x;
    const float* yr = y + (size_t)row * CDIM;
    float v0 = yr[tid], v1 = yr[tid + 256], v2 = yr[tid + 512];
    red[tid] = v0 + v1 + v2;
    __syncthreads();
    for (int o = 128; o > 0; o >>= 1) { if (tid < o) red[tid] += red[tid + o]; __syncthreads(); }
    const float mean = red[0] * (1.0f / (float)CDIM);
    __syncthreads();
    float d0 = v0 - mean, d1 = v1 - mean, d2 = v2 - mean;
    red[tid] = d0 * d0 + d1 * d1 + d2 * d2;
    __syncthreads();
    for (int o = 128; o > 0; o >>= 1) { if (tid < o) red[tid] += red[tid + o]; __syncthreads(); }
    const float rstd = rsqrtf(red[0] * (1.0f / (float)CDIM) + 1e-5f);
    const float* srr = sr + (size_t)row * CDIM;
    ushort_t* zr = zout + (size_t)row * CDIM;
    zr[tid]       = f2bf((d0 * rstd * g[tid]       + bln[tid])       * srr[tid]);
    zr[tid + 256] = f2bf((d1 * rstd * g[tid + 256] + bln[tid + 256]) * srr[tid + 256]);
    zr[tid + 512] = f2bf((d2 * rstd * g[tid + 512] + bln[tid + 512]) * srr[tid + 512]);
}

// ---------------------------------------------------------------------------
// Launch
// ---------------------------------------------------------------------------
extern "C" void kernel_launch(void* const* d_in, const int* in_sizes, int n_in,
                              void* d_out, int out_size, void* d_ws, size_t ws_size,
                              hipStream_t stream) {
    const float* x             = (const float*)d_in[0];
    const float* spatial_decay = (const float*)d_in[1];
    const float* spatial_first = (const float*)d_in[2];
    const float* mix_k         = (const float*)d_in[3];
    const float* mix_v         = (const float*)d_in[4];
    const float* mix_r         = (const float*)d_in[5];
    const float* Wk            = (const float*)d_in[6];
    const float* Wv            = (const float*)d_in[7];
    const float* Wr            = (const float*)d_in[8];
    const float* Wo            = (const float*)d_in[9];
    const float* ln_g          = (const float*)d_in[10];
    const float* ln_b          = (const float*)d_in[11];
    const float* Wd1           = (const float*)d_in[12];
    const float* bd1           = (const float*)d_in[13];
    const float* Wd2           = (const float*)d_in[14];
    const float* bd2           = (const float*)d_in[15];
    float* out                 = (float*)d_out;

    // ---- workspace layout (256B aligned regions) ----
    char* ws = (char*)d_ws;
    size_t off = 0;
    auto carve = [&](size_t bytes) { size_t r = off; off = (off + bytes + 255) & ~(size_t)255; return r; };
    const size_t per_w   = (size_t)CDIM * CDIM;                  // 589824 elems
    const size_t off_WT  = carve(4 * per_w * sizeof(ushort_t));  // [mat][n][k] bf16
    const size_t off_gc  = carve((size_t)BDIM * CDIM * sizeof(float));
    const size_t off_w   = carve((size_t)CDIM * sizeof(float));
    const size_t nloc    = (size_t)NCH * BDIM * CDIM;
    const size_t off_la  = carve(nloc * sizeof(float));
    const size_t off_lb  = carve(nloc * sizeof(float));
    const size_t off_lp  = carve(nloc * sizeof(float));
    const size_t off_sa  = carve(nloc * sizeof(float));
    const size_t off_sb  = carve(nloc * sizeof(float));
    const size_t off_sp  = carve(nloc * sizeof(float));
    const size_t mc      = (size_t)MDIM * CDIM;                  // 25,165,824 elems
    const size_t off_X   = carve(3 * mc * sizeof(ushort_t));     // Xk|Xv|Xr bf16
    const size_t off_kvs = carve(3 * mc * sizeof(float));        // k|v|sr f32

    ushort_t* WT   = (ushort_t*)(ws + off_WT);
    float*    gctx = (float*)(ws + off_gc);
    float*    wvec = (float*)(ws + off_w);
    float*    la   = (float*)(ws + off_la);
    float*    lb   = (float*)(ws + off_lb);
    float*    lp   = (float*)(ws + off_lp);
    float*    sa   = (float*)(ws + off_sa);
    float*    sb   = (float*)(ws + off_sb);
    float*    sp   = (float*)(ws + off_sp);
    ushort_t* Xk   = (ushort_t*)(ws + off_X);
    ushort_t* Xv   = Xk + mc;
    ushort_t* Xr   = Xv + mc;
    float*    kbuf = (float*)(ws + off_kvs);                     // z=0:k, z=1:v, z=2:sr
    float*    vbuf = kbuf + mc;
    float*    srb  = vbuf + mc;
    // aliases over regions that are dead by the time they are written:
    float*    ybuf = (float*)(ws + off_X);                       // over Xk+Xv (dead after GEMM1)
    ushort_t* zbuf = (ushort_t*)((char*)(ws + off_X) + mc * sizeof(float)); // over Xr

    // 1) gctx
    zero_f32_kernel<<<(BDIM * CDIM + 255) / 256, 256, 0, stream>>>(gctx, BDIM * CDIM);
    gctx_kernel<<<dim3(BDIM, 64), 256, 0, stream>>>(x, gctx);
    // 2) adaptive decay vector w
    decay_kernel<<<1, 256, 0, stream>>>(gctx, Wd1, bd1, Wd2, bd2, spatial_decay, wvec);
    // 3) weights -> bf16 transposed [n][k]
    wconv_kernel<<<(int)((4 * per_w + 255) / 256), 256, 0, stream>>>(Wk, Wv, Wr, Wo, WT);
    // 4) q_shift + mix -> bf16 operands
    qshift_mix_kernel<<<(int)((mc + 255) / 256), 256, 0, stream>>>(x, mix_k, mix_v, mix_r, Xk, Xv, Xr);
    // 5) k, v, sr via WMMA GEMM (z = 0,1,2; sigmoid on z==2)
    gemm_bf16_kernel<<<dim3(MDIM / GBM, CDIM / GBN, 3), 256, 0, stream>>>(
        Xk, WT, kbuf, MDIM, CDIM, CDIM, /*sig_z=*/2);
    // 6) WKV chunked scan
    wkv_passA<<<(int)(nloc / 256), 256, 0, stream>>>(kbuf, vbuf, wvec, la, lb, lp);
    wkv_passB<<<(BDIM * CDIM + 255) / 256, 256, 0, stream>>>(la, lb, lp, wvec, sa, sb, sp);
    wkv_passC<<<(int)(nloc / 256), 256, 0, stream>>>(kbuf, vbuf, wvec, spatial_first,
                                                     sa, sb, sp, ybuf);
    // 7) z = bf16(sr * LN(y))
    ln_kernel<<<MDIM, 256, 0, stream>>>(ybuf, srb, ln_g, ln_b, zbuf);
    // 8) out = z @ Wo  (WT mat 3)
    gemm_bf16_kernel<<<dim3(MDIM / GBM, CDIM / GBN, 1), 256, 0, stream>>>(
        zbuf, WT + 3 * per_w, out, MDIM, CDIM, CDIM, /*sig_z=*/-1);
}
